// AttModel_59596966199389
// MI455X (gfx1250) — matrix operand order repaired and checked by reading, not compiled, not run on hardware
//
#include <hip/hip_runtime.h>
#include <hip/hip_bf16.h>

typedef __attribute__((ext_vector_type(16))) _Float16 v16h;
typedef __attribute__((ext_vector_type(8)))  float    v8f;
typedef __attribute__((ext_vector_type(4)))  unsigned int u32x4;
typedef __attribute__((ext_vector_type(4)))  int          i32x4;

union FragH { v16h h; u32x4 u[2]; };

#if __has_builtin(__builtin_amdgcn_global_load_async_to_lds_b128) && \
    __has_builtin(__builtin_amdgcn_s_wait_asynccnt)
#define USE_ASYNC_LDS 1
typedef __attribute__((address_space(1))) i32x4 gas_i32x4;
typedef __attribute__((address_space(3))) i32x4 las_i32x4;
#else
#define USE_ASYNC_LDS 0
#endif

// copy 16B global -> LDS (async DMA when available, else VGPR round-trip)
__device__ __forceinline__ void copy16(const _Float16* g, _Float16* l) {
#if USE_ASYNC_LDS
    __builtin_amdgcn_global_load_async_to_lds_b128(
        (gas_i32x4*)(_Float16*)g, (las_i32x4*)l, 0, 0);
#else
    *reinterpret_cast<u32x4*>(l) = *reinterpret_cast<const u32x4*>(g);
#endif
}
__device__ __forceinline__ void async_wait() {
#if USE_ASYNC_LDS
    __builtin_amdgcn_s_wait_asynccnt(0);
#endif
}

__device__ __forceinline__ float act_f(float x, int act) {
    if (act == 1) return x > 0.f ? x : 0.f;
    if (act == 2) return 1.f / (1.f + __expf(-x));
    if (act == 3) return tanhf(x);
    return x;
}

__device__ __forceinline__ float wave_reduce(float s) {
    #pragma unroll
    for (int off = 16; off >= 1; off >>= 1) s += __shfl_xor(s, off, 32);
    return s;
}

// ---------------------------------------------------------------------------
// zero helper
__global__ void zero_f32(float* p, int n) {
    int i = blockIdx.x * 256 + threadIdx.x;
    if (i < n) p[i] = 0.f;
}

// ---------------------------------------------------------------------------
// pack OIHW f32 weights -> [Cout][Kh][Kw][Cin] f16  (Kh=Kw=1 => plain convert)
__global__ void pack_w(const float* __restrict__ w, _Float16* __restrict__ out,
                       int Cout, int Cin, int Kh, int Kw) {
    int idx = blockIdx.x * 256 + threadIdx.x;
    int total = Cout * Cin * Kh * Kw;
    if (idx >= total) return;
    int ci = idx % Cin; int r = idx / Cin;
    int kw = r % Kw; r /= Kw;
    int kh = r % Kh; int co = r / Kh;
    out[idx] = (_Float16)w[((size_t)(co * Cin + ci) * Kh + kh) * Kw + kw];
}

// ---------------------------------------------------------------------------
// conv1: [3,512,512] f32 -> [255,255,128] HWC f16, 3x3 stride 2 VALID
__global__ void conv1_direct(const float* __restrict__ x, const float* __restrict__ w,
                             const float* __restrict__ b, _Float16* __restrict__ out) {
    int idx = blockIdx.x * 256 + threadIdx.x;
    if (idx >= 255 * 255 * 128) return;
    int co = idx & 127;
    int p  = idx >> 7;
    int y = p / 255, xx = p % 255;
    float acc = b[co];
    #pragma unroll
    for (int ci = 0; ci < 3; ++ci)
        #pragma unroll
        for (int kh = 0; kh < 3; ++kh)
            #pragma unroll
            for (int kw = 0; kw < 3; ++kw)
                acc += x[ci * 512 * 512 + (2 * y + kh) * 512 + (2 * xx + kw)] *
                       w[((co * 3 + ci) * 3 + kh) * 3 + kw];
    out[(size_t)p * 128 + co] = (_Float16)acc;
}

// ---------------------------------------------------------------------------
// Implicit-GEMM conv via v_wmma_f32_16x16x32_f16, double-buffered LDS with
// async global->LDS staging. Input HWC f16, weights [N][K] f16
// (K = (kh*Kw+kw)*Cin+ci), output HWC f16.
// Block tile: 64(M) x 128(N), K-step 64. 8 waves, each owns a 32x32 sub-tile
// (4 accumulators) -> 8 WMMAs per K-chunk per wave per barrier.
// Requires: Cin % 64 == 0, Cout % 128 == 0.
// act: 0 none, 1 relu, 2 sigmoid, 3 tanh, 4 (sigmoid if co<256 else tanh).
__global__ __launch_bounds__(256)
void conv_wmma(const _Float16* __restrict__ in, const _Float16* __restrict__ wpack,
               const float* __restrict__ bias, _Float16* __restrict__ out,
               int Hout, int Wout, int Win, int Cin, int Cout,
               int Kh, int Kw, int stride, int act) {
    __shared__ __align__(16) _Float16 As[2][64 * 64];    // 2 x 8 KB
    __shared__ __align__(16) _Float16 Bs[2][128 * 64];   // 2 x 16 KB

    const int M    = Hout * Wout;
    const int Ktot = Kh * Kw * Cin;
    const int mbase = blockIdx.x * 64;
    const int nbase = blockIdx.y * 128;
    const int tid  = threadIdx.x;
    const int lane = tid & 31;
    const int wid  = tid >> 5;
    const int mw   = wid & 1;   // M half: rows mw*32 + {0,16}
    const int nw   = wid >> 1;  // N quarter: cols nw*32 + {0,16}

    v8f accv[2][2] = {};

    // staging indices
    const int arow = tid >> 2;  // 0..63, 32B (2 x 16B) per thread
    const int aseg = tid & 3;
    const int brow = tid >> 1;  // 0..127, 64B (4 x 16B) per thread
    const int bseg = tid & 1;

    const int pA = mbase + arow;
    const bool aValid = pA < M;
    int ay = 0, ax = 0;
    if (aValid) { ay = pA / Wout; ax = pA % Wout; }

    const int cdiv    = Cin >> 6;          // K-chunks per (kh,kw) slice
    const int nchunks = Kh * Kw * cdiv;

    auto stage = [&](int ck, int buf) {
        const int kc = (ck % cdiv) << 6;
        const int r  = ck / cdiv;
        const int kw = r % Kw, kh = r / Kw;
        const int kidx = (kh * Kw + kw) * Cin + kc;
        // A tile 64x64 (im2col gather; 128B contiguous per row)
        _Float16* al = As[buf] + arow * 64 + aseg * 16;
        if (aValid) {
            const _Float16* g = in + ((size_t)(ay * stride + kh) * Win +
                                      (ax * stride + kw)) * Cin + kc + aseg * 16;
            copy16(g, al);
            copy16(g + 8, al + 8);
        } else {
            *reinterpret_cast<u32x4*>(al)     = (u32x4)0;
            *reinterpret_cast<u32x4*>(al + 8) = (u32x4)0;
        }
        // B tile 128x64
        const _Float16* bp = wpack + (size_t)(nbase + brow) * Ktot + kidx + bseg * 32;
        _Float16* bl = Bs[buf] + brow * 64 + bseg * 32;
        #pragma unroll
        for (int q = 0; q < 4; ++q)
            copy16(bp + q * 8, bl + q * 8);
        // prefetch weight stream one more chunk ahead (global_prefetch_b8)
        if (ck + 1 < nchunks) {
            const int kc2 = ((ck + 1) % cdiv) << 6;
            const int r2  = (ck + 1) / cdiv;
            const int kidx2 = ((r2 / Kw) * Kw + (r2 % Kw)) * Cin + kc2;
            __builtin_prefetch(wpack + (size_t)(nbase + brow) * Ktot + kidx2 + bseg * 32, 0, 0);
        }
    };

    // prologue: fill buffer 0
    stage(0, 0);
    async_wait();
    __syncthreads();

    for (int ck = 0; ck < nchunks; ++ck) {
        const int cur = ck & 1;
        if (ck + 1 < nchunks) stage(ck + 1, cur ^ 1);

        const _Float16* Ac = As[cur];
        const _Float16* Bc = Bs[cur];
        // ---- compute: 2 K-halves x (2 A frags x 2 B frags) WMMAs ----
        #pragma unroll
        for (int kk = 0; kk < 2; ++kk) {
            const int kof = kk * 32;
            FragH a[2], b[2];
            const int ka0 = kof + ((lane < 16) ? 0 : 8);
            const int kb0 = kof + ((lane < 16) ? 0 : 8);
            #pragma unroll
            for (int i = 0; i < 2; ++i) {
                const int m = mw * 32 + i * 16 + (lane & 15);
                a[i].u[0] = *reinterpret_cast<const u32x4*>(Ac + m * 64 + ka0);
                a[i].u[1] = *reinterpret_cast<const u32x4*>(Ac + m * 64 + ka0 + 16);
            }
            #pragma unroll
            for (int j = 0; j < 2; ++j) {
                const int n = nw * 32 + j * 16 + (lane & 15);
                b[j].u[0] = *reinterpret_cast<const u32x4*>(Bc + n * 64 + kb0);
                b[j].u[1] = *reinterpret_cast<const u32x4*>(Bc + n * 64 + kb0 + 16);
            }
            #pragma unroll
            for (int i = 0; i < 2; ++i)
                #pragma unroll
                for (int j = 0; j < 2; ++j)
                    accv[i][j] = __builtin_amdgcn_wmma_f32_16x16x32_f16(
                        false, a[i].h, false, b[j].h, (short)0, accv[i][j], false, false);
        }
        // make next buffer's async DMA writes visible to the whole workgroup
        async_wait();
        __syncthreads();
    }

    // ---- epilogue: bias + activation + store ----
    #pragma unroll
    for (int i = 0; i < 2; ++i)
        #pragma unroll
        for (int j = 0; j < 2; ++j) {
            const v8f acc = accv[i][j];
            const int co = nbase + nw * 32 + j * 16 + (lane & 15);
            const float bsv = bias[co];
            int a2 = act;
            if (act == 4) a2 = (co & 256) ? 3 : 2;   // fused U|V GEMM (Cout==512)
            #pragma unroll
            for (int rr = 0; rr < 8; ++rr) {
                const int m = mw * 32 + i * 16 + ((lane < 16) ? rr : (8 + rr));
                const int p = mbase + m;
                if (p < M)
                    out[(size_t)p * Cout + co] = (_Float16)act_f(acc[rr] + bsv, a2);
            }
        }
}

// ---------------------------------------------------------------------------
// att[i] = sigmoid(sum_ch u*v*Wa + b); u = uv[:, :256], v = uv[:, 256:]
__global__ void att_kernel(const _Float16* __restrict__ uv,
                           const float* __restrict__ waw, const float* __restrict__ wab,
                           float* __restrict__ att, int N) {
    int wid = threadIdx.x >> 5, lane = threadIdx.x & 31;
    int i = blockIdx.x * 8 + wid;
    if (i >= N) return;
    float s = 0.f;
    for (int ch = lane; ch < 256; ch += 32)
        s += (float)uv[(size_t)i * 512 + ch] * (float)uv[(size_t)i * 512 + 256 + ch] * waw[ch];
    s = wave_reduce(s);
    if (lane == 0) att[i] = 1.f / (1.f + __expf(-(s + wab[0])));
}

// ---------------------------------------------------------------------------
// exact descending rank (stable: ties broken by index), cid = rank*10 / N
__global__ void rank_kernel(const float* __restrict__ att, int* __restrict__ cid, int N) {
    __shared__ float tile[256];
    int i = blockIdx.x * 256 + threadIdx.x;
    float ai = (i < N) ? att[i] : 0.f;
    int cnt = 0;
    for (int jb = 0; jb < N; jb += 256) {
        int j = jb + threadIdx.x;
        tile[threadIdx.x] = (j < N) ? att[j] : 0.f;
        __syncthreads();
        int lim = (N - jb) < 256 ? (N - jb) : 256;
        for (int t = 0; t < lim; ++t) {
            float aj = tile[t];
            int j2 = jb + t;
            cnt += ((aj > ai) || (aj == ai && j2 < i)) ? 1 : 0;
        }
        __syncthreads();
    }
    if (i < N) cid[i] = (cnt * 10) / N;
}

// ---------------------------------------------------------------------------
// coords of anchor i (computed on the fly, Hf=Wf=63)
__device__ __forceinline__ float coord_of(int i, int comp) {
    int off, Wa; float kh, kw;
    if      (i < 3721)  { off = 0;     Wa = 61; kh = 3.f; kw = 3.f; }
    else if (i < 7690)  { off = 3721;  Wa = 63; kh = 1.f; kw = 1.f; }
    else if (i < 11533) { off = 7690;  Wa = 63; kh = 3.f; kw = 1.f; }
    else                { off = 11533; Wa = 61; kh = 1.f; kw = 3.f; }
    int p = i - off;
    int ys = p / Wa, xs = p % Wa;
    const float F = 63.f;
    switch (comp) {
        case 0:  return (xs + kw * 0.5f) / F;
        case 1:  return (ys + kh * 0.5f) / F;
        case 2:  return kw / F;
        default: return kh / F;
    }
}

// attention-weighted segment sums into 10 clusters (LDS accum + global flush)
__global__ __launch_bounds__(256)
void segment_kernel(const _Float16* __restrict__ hmid, const float* __restrict__ att,
                    const int* __restrict__ cid, float* __restrict__ fc_sum,
                    float* __restrict__ sw_sum, float* __restrict__ pc_sum, int N) {
    __shared__ float fcl[10 * 512];
    __shared__ float swl[10];
    __shared__ float pcl[40];
    int tid = threadIdx.x;
    for (int t = tid; t < 10 * 512; t += 256) fcl[t] = 0.f;
    if (tid < 10) swl[tid] = 0.f;
    if (tid < 40) pcl[tid] = 0.f;
    __syncthreads();
    int wid = tid >> 5, lane = tid & 31;
    int base = blockIdx.x * 256;
    for (int it = 0; it < 32; ++it) {
        int i = base + it * 8 + wid;
        if (i < N) {
            int c = cid[i];
            float a = att[i];
            for (int ch = lane; ch < 512; ch += 32)
                atomicAdd(&fcl[c * 512 + ch], (float)hmid[(size_t)i * 512 + ch] * a);
            if (lane == 0) atomicAdd(&swl[c], a);
            if (lane < 4)  atomicAdd(&pcl[c * 4 + lane], coord_of(i, lane) * a);
        }
    }
    __syncthreads();
    for (int t = tid; t < 10 * 512; t += 256)
        if (fcl[t] != 0.f) atomicAdd(&fc_sum[t], fcl[t]);
    if (tid < 10) atomicAdd(&sw_sum[tid], swl[tid]);
    if (tid < 40) atomicAdd(&pc_sum[tid], pcl[tid]);
}

// ---------------------------------------------------------------------------
// cluster attention + pooling + classifier; writes all 60 outputs
__global__ __launch_bounds__(512)
void final_kernel(const float* __restrict__ fc_sum, const float* __restrict__ sw_sum,
                  const float* __restrict__ pc_sum, const float* __restrict__ cluster_w,
                  const float* __restrict__ cluster_b, const float* __restrict__ cls_w,
                  const float* __restrict__ cls_b, float* __restrict__ out) {
    __shared__ float scores[16];
    __shared__ float attc[10];
    __shared__ float finalv[512];
    __shared__ float logits[16];
    int tid = threadIdx.x, wid = tid >> 5, lane = tid & 31;
    if (wid < 10) {
        float inv = 1.f / (sw_sum[wid] + 1e-8f);
        float s = 0.f;
        for (int ch = lane; ch < 512; ch += 32)
            s += fc_sum[wid * 512 + ch] * inv * cluster_w[ch];
        s = wave_reduce(s);
        if (lane == 0) scores[wid] = s + cluster_b[0];
    }
    __syncthreads();
    if (tid == 0) {
        float mx = -1e30f;
        for (int c = 0; c < 10; ++c) mx = fmaxf(mx, scores[c]);
        float sum = 0.f;
        for (int c = 0; c < 10; ++c) { attc[c] = __expf(scores[c] - mx); sum += attc[c]; }
        for (int c = 0; c < 10; ++c) attc[c] /= sum;
    }
    __syncthreads();
    {
        float f = 0.f;
        for (int c = 0; c < 10; ++c)
            f += attc[c] * fc_sum[c * 512 + tid] / (sw_sum[c] + 1e-8f);
        finalv[tid] = f;
    }
    __syncthreads();
    if (wid < 10) {
        float s = 0.f;
        for (int ch = lane; ch < 512; ch += 32)
            s += finalv[ch] * cls_w[wid * 512 + ch];
        s = wave_reduce(s);
        if (lane == 0) logits[wid] = s + cls_b[wid];
    }
    __syncthreads();
    if (tid == 0) {
        float mx = -1e30f;
        for (int c = 0; c < 10; ++c) mx = fmaxf(mx, logits[c]);
        float e[10]; float sum = 0.f;
        for (int c = 0; c < 10; ++c) { e[c] = __expf(logits[c] - mx); sum += e[c]; }
        for (int c = 0; c < 10; ++c) out[c] = e[c] / sum;        // softmax(logits)
        for (int c = 0; c < 10; ++c) out[10 + c] = attc[c];      // att_c
    }
    if (tid < 40) out[20 + tid] = pc_sum[tid] / (sw_sum[tid >> 2] + 1e-8f);  // pc
}

// ---------------------------------------------------------------------------
extern "C" void kernel_launch(void* const* d_in, const int* in_sizes, int n_in,
                              void* d_out, int out_size, void* d_ws, size_t ws_size,
                              hipStream_t stream) {
    (void)in_sizes; (void)n_in; (void)out_size; (void)ws_size;
    const float* x         = (const float*)d_in[0];
    const float* conv1_w   = (const float*)d_in[1];  const float* conv1_b = (const float*)d_in[2];
    const float* conv2_w   = (const float*)d_in[3];  const float* conv2_b = (const float*)d_in[4];
    const float* conv3_w   = (const float*)d_in[5];  const float* conv3_b = (const float*)d_in[6];
    const float* ancw[4]   = { (const float*)d_in[7],  (const float*)d_in[9],
                               (const float*)d_in[11], (const float*)d_in[13] };
    const float* ancb[4]   = { (const float*)d_in[8],  (const float*)d_in[10],
                               (const float*)d_in[12], (const float*)d_in[14] };
    const float* W1_w = (const float*)d_in[15]; const float* W1_b = (const float*)d_in[16];
    const float* U_w  = (const float*)d_in[17]; const float* U_b  = (const float*)d_in[18];
    const float* V_w  = (const float*)d_in[19]; const float* V_b  = (const float*)d_in[20];
    const float* Wa_w = (const float*)d_in[21]; const float* Wa_b = (const float*)d_in[22];
    const float* cluster_w = (const float*)d_in[23]; const float* cluster_b = (const float*)d_in[24];
    const float* cls_w = (const float*)d_in[25]; const float* cls_b = (const float*)d_in[26];
    float* out = (float*)d_out;

    // ---- workspace layout (256B aligned slices) ----
    char* ws = (char*)d_ws;
    size_t off = 0;
    auto alloc = [&](size_t bytes) -> void* {
        void* p = ws + off;
        off += (bytes + 255) & ~size_t(255);
        return p;
    };
    const int N = 15376;
    _Float16* h1   = (_Float16*)alloc((size_t)255 * 255 * 128 * 2);
    _Float16* h2   = (_Float16*)alloc((size_t)127 * 127 * 256 * 2);
    _Float16* h3   = (_Float16*)alloc((size_t)63 * 63 * 512 * 2);
    _Float16* feat = (_Float16*)alloc((size_t)N * 1024 * 2);
    _Float16* hmid = (_Float16*)alloc((size_t)N * 512 * 2);
    _Float16* uvb  = (_Float16*)alloc((size_t)N * 512 * 2);
    float*    att  = (float*)alloc((size_t)N * 4);
    int*      cid  = (int*)alloc((size_t)N * 4);
    _Float16* wp2  = (_Float16*)alloc((size_t)256 * 1152 * 2);
    _Float16* wp3  = (_Float16*)alloc((size_t)512 * 2304 * 2);
    _Float16* wpa0 = (_Float16*)alloc((size_t)1024 * 4608 * 2);
    _Float16* wpa1 = (_Float16*)alloc((size_t)1024 * 512 * 2);
    _Float16* wpa2 = (_Float16*)alloc((size_t)1024 * 1536 * 2);
    _Float16* wpa3 = (_Float16*)alloc((size_t)1024 * 1536 * 2);
    _Float16* wpW1 = (_Float16*)alloc((size_t)512 * 1024 * 2);
    _Float16* wpUV = (_Float16*)alloc((size_t)512 * 512 * 2);
    float* uv_b    = (float*)alloc(512 * 4);
    float* fc_sum  = (float*)alloc(10 * 512 * 4);
    float* sw_sum  = (float*)alloc(10 * 4);
    float* pc_sum  = (float*)alloc(40 * 4);

    // ---- zero accumulators (harness does not re-zero between replays) ----
    zero_f32<<<(10 * 512 + 255) / 256, 256, 0, stream>>>(fc_sum, 10 * 512);
    zero_f32<<<1, 256, 0, stream>>>(sw_sum, 10);
    zero_f32<<<1, 256, 0, stream>>>(pc_sum, 40);

    // ---- pack weights to f16 [N][K] ----
    auto pack = [&](const float* w, _Float16* o, int Co, int Ci, int Kh, int Kw) {
        int total = Co * Ci * Kh * Kw;
        pack_w<<<(total + 255) / 256, 256, 0, stream>>>(w, o, Co, Ci, Kh, Kw);
    };
    pack(conv2_w, wp2, 256, 128, 3, 3);
    pack(conv3_w, wp3, 512, 256, 3, 3);
    pack(ancw[0], wpa0, 1024, 512, 3, 3);
    pack(ancw[1], wpa1, 1024, 512, 1, 1);
    pack(ancw[2], wpa2, 1024, 512, 3, 1);
    pack(ancw[3], wpa3, 1024, 512, 1, 3);
    pack(W1_w, wpW1, 512, 1024, 1, 1);
    pack(U_w, wpUV,              256, 512, 1, 1);   // rows   0..255
    pack(V_w, wpUV + 256 * 512,  256, 512, 1, 1);   // rows 256..511
    (void)hipMemcpyAsync(uv_b,       U_b, 256 * sizeof(float), hipMemcpyDeviceToDevice, stream);
    (void)hipMemcpyAsync(uv_b + 256, V_b, 256 * sizeof(float), hipMemcpyDeviceToDevice, stream);

    // ---- backbone ----
    conv1_direct<<<(255 * 255 * 128 + 255) / 256, 256, 0, stream>>>(x, conv1_w, conv1_b, h1);

    auto conv = [&](const _Float16* in, const _Float16* wp, const float* bias, _Float16* o,
                    int Ho, int Wo, int Wi, int Ci, int Co, int Kh, int Kw, int s, int act) {
        dim3 g((Ho * Wo + 63) / 64, Co / 128);
        conv_wmma<<<g, 256, 0, stream>>>(in, wp, bias, o, Ho, Wo, Wi, Ci, Co, Kh, Kw, s, act);
    };
    conv(h1, wp2, conv2_b, h2, 127, 127, 255, 128, 256, 3, 3, 2, 0);
    conv(h2, wp3, conv3_b, h3,  63,  63, 127, 256, 512, 3, 3, 2, 0);

    // ---- anchors -> feat slices (concat order anc0..anc3, row-major pixels) ----
    conv(h3, wpa0, ancb[0], feat + (size_t)0     * 1024, 61, 61, 63, 512, 1024, 3, 3, 1, 0);
    conv(h3, wpa1, ancb[1], feat + (size_t)3721  * 1024, 63, 63, 63, 512, 1024, 1, 1, 1, 0);
    conv(h3, wpa2, ancb[2], feat + (size_t)7690  * 1024, 61, 63, 63, 512, 1024, 3, 1, 1, 0);
    conv(h3, wpa3, ancb[3], feat + (size_t)11533 * 1024, 63, 61, 63, 512, 1024, 1, 3, 1, 0);

    // ---- attention head GEMMs (as 1x1 "conv", Hout=1, Wout=N) ----
    conv(feat, wpW1, W1_b, hmid, 1, N, N, 1024, 512, 1, 1, 1, 1); // relu
    conv(hmid, wpUV, uv_b, uvb,  1, N, N, 512,  512, 1, 1, 1, 4); // fused sigmoid|tanh

    // ---- attention scores, ranking, clustering ----
    att_kernel<<<(N + 7) / 8, 256, 0, stream>>>(uvb, Wa_w, Wa_b, att, N);
    rank_kernel<<<(N + 255) / 256, 256, 0, stream>>>(att, cid, N);
    segment_kernel<<<(N + 255) / 256, 256, 0, stream>>>(hmid, att, cid, fc_sum, sw_sum, pc_sum, N);

    // ---- cluster attention + classifier -> 60 outputs ----
    final_kernel<<<1, 512, 0, stream>>>(fc_sum, sw_sum, pc_sum, cluster_w, cluster_b,
                                        cls_w, cls_b, out);
}
